// VectorQuantizer_65231963292130
// MI455X (gfx1250) — compile-verified, matrix operand order checked by reading, not run on hardware
//
#include <hip/hip_runtime.h>

// Vector quantizer for MI455X (gfx1250).
// Compute-bound: 32768x8192x256 GEMM (137 GFLOP) vs ~75MB traffic (3us @ 23.3TB/s).
// Strategy: split-bf16 (z=zH+zL, e=eH+eL) -> 3x v_wmma_f32_16x16x32_bf16 per
// K-chunk with f32 accumulation; argmin fused in registers; tiny prep/gather passes.

typedef __attribute__((ext_vector_type(16))) __bf16 v16bf;
typedef __attribute__((ext_vector_type(8)))  float  v8f;
typedef unsigned short ushort_t;

#define N_E   8192
#define E_DIM 256
#define BATCH 32
#define HW    1024          // 32*32
#define N_Z   32768         // BATCH*HW
#define ZQ_ELEMS 8388608    // 32*256*32*32

__device__ __forceinline__ ushort_t f2bf(float f) {
    unsigned u = __float_as_uint(f);
    unsigned r = u + 0x7FFFu + ((u >> 16) & 1u);   // round-to-nearest-even
    return (ushort_t)(r >> 16);
}
__device__ __forceinline__ float bf2f(ushort_t h) {
    return __uint_as_float(((unsigned)h) << 16);
}

// ---------------- Kernel 1: emb -> (eH, eL) bf16 + esq = ||e||^2 -------------
__global__ __launch_bounds__(256) void prep_emb_kernel(
    const float* __restrict__ emb, ushort_t* __restrict__ eH,
    ushort_t* __restrict__ eL, float* __restrict__ esq)
{
    __shared__ float red[256];
    const int row = blockIdx.x;          // 0..8191
    const int t   = threadIdx.x;         // 0..255
    const float v = emb[(size_t)row * E_DIM + t];
    const ushort_t h = f2bf(v);
    const float    hf = bf2f(h);
    const ushort_t l = f2bf(v - hf);
    eH[(size_t)row * E_DIM + t] = h;
    eL[(size_t)row * E_DIM + t] = l;
    red[t] = v * v;
    __syncthreads();
    for (int o = 128; o > 0; o >>= 1) {
        if (t < o) red[t] += red[t + o];
        __syncthreads();
    }
    if (t == 0) esq[row] = red[0];
}

// ---------------- Kernel 2: z NCHW -> z_flat NHWC, bf16 hi/lo split ----------
// LDS-tiled 32x32 transpose per (b, c-tile, hw-tile); coalesced both directions.
__global__ __launch_bounds__(256) void prep_z_kernel(
    const float* __restrict__ z, ushort_t* __restrict__ zH,
    ushort_t* __restrict__ zL)
{
    __shared__ float tile[32][33];
    const int b   = blockIdx.z;
    const int c0  = blockIdx.y * 32;
    const int hw0 = blockIdx.x * 32;
    const int tx  = threadIdx.x;          // 0..31
    const int ty  = threadIdx.y;          // 0..7
    const float* zp = z + (size_t)b * (E_DIM * HW);
    #pragma unroll
    for (int i = 0; i < 32; i += 8) {
        const int c = c0 + ty + i;
        tile[ty + i][tx] = zp[(size_t)c * HW + hw0 + tx];   // coalesced over hw
    }
    __syncthreads();
    #pragma unroll
    for (int i = 0; i < 32; i += 8) {
        const int hw = hw0 + ty + i;
        const size_t n = (size_t)b * HW + hw;
        const float v = tile[tx][ty + i];                   // [c_local][hw_local]
        const ushort_t h = f2bf(v);
        const float    hf = bf2f(h);
        const ushort_t l = f2bf(v - hf);
        zH[n * E_DIM + c0 + tx] = h;                        // coalesced over c
        zL[n * E_DIM + c0 + tx] = l;
    }
}

// ---------------- Kernel 3: fused GEMM + argmin ------------------------------
// Wave owns 16 z-rows; A(hi+lo, K=256) resident in 128 VGPRs.
// A/B K order used here (contiguous): lane l<16 holds row (l), K chunk kk..kk+15;
// lane l+16 holds row (l), K kk+16..kk+31  -> one 32B contiguous load per lane.
__global__ __launch_bounds__(256) void vq_argmin_kernel(
    const ushort_t* __restrict__ zH, const ushort_t* __restrict__ zL,
    const ushort_t* __restrict__ eH, const ushort_t* __restrict__ eL,
    const float* __restrict__ esq, int* __restrict__ idx_out,
    float* __restrict__ idxf_out)
{
    const int lane = threadIdx.x & 31;
    const int wave = threadIdx.x >> 5;
    const int n0   = (blockIdx.x * 8 + wave) * 16;
    const int rl   = lane & 15;        // row (A) / col (B) within tile
    const int khalf = lane >> 4;       // which 16-wide K half this lane holds

    // Resident A tiles (hi and lo), 8 K-chunks of 32.
    v16bf aH[8], aL[8];
    {
        const size_t abase = ((size_t)(n0 + rl)) * E_DIM + (khalf << 4);
        #pragma unroll
        for (int kk = 0; kk < 8; ++kk) {
            union { uint4 u[2]; v16bf v; } tH, tL;
            const uint4* pH = (const uint4*)(zH + abase + kk * 32);
            const uint4* pL = (const uint4*)(zL + abase + kk * 32);
            tH.u[0] = pH[0]; tH.u[1] = pH[1];
            tL.u[0] = pL[0]; tL.u[1] = pL[1];
            aH[kk] = tH.v;   aL[kk] = tL.v;
        }
    }

    float best[8];
    int   bidx[8];
    #pragma unroll
    for (int k = 0; k < 8; ++k) { best[k] = 3.4e38f; bidx[k] = 0; }

    for (int m0 = 0; m0 < N_E; m0 += 16) {
        v8f acc = {};
        const size_t bbase = ((size_t)(m0 + rl)) * E_DIM + (khalf << 4);
        #pragma unroll
        for (int kk = 0; kk < 8; ++kk) {
            union { uint4 u[2]; v16bf v; } tH, tL;
            const uint4* pH = (const uint4*)(eH + bbase + kk * 32);
            const uint4* pL = (const uint4*)(eL + bbase + kk * 32);
            tH.u[0] = pH[0]; tH.u[1] = pH[1];
            tL.u[0] = pL[0]; tL.u[1] = pL[1];
            // zH*eH + zH*eL + zL*eH  (f32 accumulate) ~ full f32 dot product
            acc = __builtin_amdgcn_wmma_f32_16x16x32_bf16(false, aH[kk], false, tH.v,
                                                          (short)0, acc, false, false);
            acc = __builtin_amdgcn_wmma_f32_16x16x32_bf16(false, aH[kk], false, tL.v,
                                                          (short)0, acc, false, false);
            acc = __builtin_amdgcn_wmma_f32_16x16x32_bf16(false, aL[kk], false, tH.v,
                                                          (short)0, acc, false, false);
        }
        // d = ||e||^2 - 2 z.e  (||z||^2 constant per row -> irrelevant to argmin)
        const int   col = m0 + rl;            // column this lane holds
        const float eq  = esq[col];
        #pragma unroll
        for (int k = 0; k < 8; ++k) {
            const float d = fmaf(-2.0f, acc[k], eq);
            if (d < best[k]) { best[k] = d; bidx[k] = col; }
        }
    }

    // Butterfly min-reduce across the 16 columns (xor 1,2,4,8 stays in-group).
    #pragma unroll
    for (int off = 1; off < 16; off <<= 1) {
        #pragma unroll
        for (int k = 0; k < 8; ++k) {
            const float ob = __shfl_xor(best[k], off, 32);
            const int   oi = __shfl_xor(bidx[k], off, 32);
            if (ob < best[k] || (ob == best[k] && oi < bidx[k])) {
                best[k] = ob; bidx[k] = oi;
            }
        }
    }
    // lanes 0-15 hold rows n0..n0+7 in acc regs 0..7; lanes 16-31 rows n0+8..15
    if (rl == 0) {
        #pragma unroll
        for (int k = 0; k < 8; ++k) {
            const int row = n0 + khalf * 8 + k;
            idx_out[row]  = bidx[k];
            idxf_out[row] = (float)bidx[k];   // indices output (exact in f32)
        }
    }
}

// ---------------- Kernel 4: gather z_q = z + (emb[idx] - z), NHWC->NCHW ------
__global__ __launch_bounds__(256) void gather_kernel(
    const float* __restrict__ z, const float* __restrict__ emb,
    const int* __restrict__ idx, float* __restrict__ out)
{
    const int b  = blockIdx.z;
    const int hw = blockIdx.x * 64 + threadIdx.x;
    const int c  = blockIdx.y * 4 + threadIdx.y;
    const int n  = b * HW + hw;
    const int e  = idx[n];                               // coalesced over hw
    const size_t o = (size_t)b * (E_DIM * HW) + (size_t)c * HW + hw;
    const float zv = z[o];
    const float ev = emb[(size_t)e * E_DIM + c];         // L2-resident gather
    out[o] = zv + (ev - zv);   // straight-through forward, matches ref rounding
}

extern "C" void kernel_launch(void* const* d_in, const int* in_sizes, int n_in,
                              void* d_out, int out_size, void* d_ws, size_t ws_size,
                              hipStream_t stream) {
    const float* z   = (const float*)d_in[0];   // (32,256,32,32) f32
    const float* emb = (const float*)d_in[1];   // (8192,256) f32
    float* out = (float*)d_out;                 // z_q (8388608) ++ indices (32768)

    // Workspace layout (42,106,880 bytes, all 256B aligned)
    char* ws = (char*)d_ws;
    ushort_t* eH  = (ushort_t*)(ws);                    //  4,194,304
    ushort_t* eL  = (ushort_t*)(ws +  4194304);         //  4,194,304
    ushort_t* zH  = (ushort_t*)(ws +  8388608);         // 16,777,216
    ushort_t* zL  = (ushort_t*)(ws + 25165824);         // 16,777,216
    float*    esq = (float*)   (ws + 41943040);         //     32,768
    int*      idx = (int*)     (ws + 41975808);         //    131,072

    prep_emb_kernel<<<N_E, 256, 0, stream>>>(emb, eH, eL, esq);
    prep_z_kernel<<<dim3(HW / 32, E_DIM / 32, BATCH), dim3(32, 8), 0, stream>>>(z, zH, zL);
    vq_argmin_kernel<<<N_Z / 128, 256, 0, stream>>>(zH, zL, eH, eL, esq,
                                                    idx, out + ZQ_ELEMS);
    gather_kernel<<<dim3(HW / 64, E_DIM / 4, BATCH), dim3(64, 4), 0, stream>>>(z, emb, idx, out);
}
